// RelationalCritic_44461501449025
// MI455X (gfx1250) — compile-verified, matrix operand order checked by reading, not run on hardware
//
#include <hip/hip_runtime.h>

#define A_ 3
#define B_ 4096
#define N_ 64
#define FEAT_ 32
#define H_ 32
#define R_ 6
#define EPER_ 1536
#define NACT_ 16

typedef __attribute__((ext_vector_type(16))) _Float16 v16h;
typedef __attribute__((ext_vector_type(8)))  float    v8f;
typedef __attribute__((ext_vector_type(4)))  unsigned int u32x4;
typedef __attribute__((ext_vector_type(8)))  int    i32x8;
typedef __attribute__((ext_vector_type(4)))  int    i32x4;

union FragAB { v16h v; uint4 q[2]; };
union FragC  { v8f  v; float f[8]; };

// ---------------------------------------------------------------------------
// TDM: 1-D contiguous tile, global -> LDS. D# per CDNA5 ISA ch.8:
//  group0: [1:0]count=1 | lds_addr | global_addr[56:0] | type=2
//  group1: data_size code | tensor_dim0/1 | tile_dim0 | tensor_dim0_stride
//  groups 2/3: zero (tensor <= 2D)
// elem_code: 0=1B 1=2B 2=4B 3=8B ; n_elems in elements (fits 16-bit tile_dim0)
// This toolchain exposes the 6-arg builtin (extra int32x8 group before cpol).
// ---------------------------------------------------------------------------
__device__ __forceinline__ void tdm_load_1d(unsigned lds_off, const void* gptr,
                                            unsigned n_elems, unsigned elem_code) {
  unsigned long long ga = (unsigned long long)(uintptr_t)gptr;
  u32x4 g0;
  g0[0] = 1u;                                            // count=1 (valid), user mode
  g0[1] = lds_off;                                       // lds_addr (bytes)
  g0[2] = (unsigned)(ga & 0xFFFFFFFFu);                  // global_addr lo
  g0[3] = (unsigned)((ga >> 32) & 0x1FFFFFFu) | (2u << 30); // addr hi | type=2
  i32x8 g1;
  g1[0] = (int)(elem_code << 16);                        // wg_mask=0 | data_size
  g1[1] = (int)((n_elems & 0xFFFFu) << 16);              // tensor_dim0[15:0]
  g1[2] = (int)((n_elems >> 16) & 0xFFFFu) | (1 << 16);  // dim0 hi | tensor_dim1=1
  g1[3] = (int)((n_elems & 0xFFFFu) << 16);              // dim1 hi(0) | tile_dim0
  g1[4] = 0;                                             // tile_dim1=0, tile_dim2=0
  g1[5] = (int)n_elems;                                  // tensor_dim0_stride lo32
  g1[6] = 0;
  g1[7] = 0;
  i32x4 z4 = {0, 0, 0, 0};
  i32x8 z8 = {0, 0, 0, 0, 0, 0, 0, 0};
  __builtin_amdgcn_tensor_load_to_lds(g0, g1, z4, z4, z8, 0);
}

// ---------------------------------------------------------------------------
// Prep kernel: per-(dst,rel) degree -> per-edge norm (identical for every
// graph since the edge list is shared), packed edge records; f16-transposed
// weights [H][K] so WMMA B-fragments are contiguous 16B chunks.
// ---------------------------------------------------------------------------
__global__ void prep_kernel(const int* __restrict__ esrc,
                            const int* __restrict__ edst,
                            const int* __restrict__ erel,
                            const float* __restrict__ w_rel,
                            const float* __restrict__ w_root,
                            uint2* __restrict__ ws_edges,
                            _Float16* __restrict__ wrelT,
                            _Float16* __restrict__ wrootT) {
  __shared__ int cnt[N_ * R_];
  const int t = threadIdx.x;
  for (int i = t; i < N_ * R_; i += blockDim.x) cnt[i] = 0;
  __syncthreads();
  for (int e = t; e < EPER_; e += blockDim.x)
    atomicAdd(&cnt[edst[e] * R_ + erel[e]], 1);
  __syncthreads();
  for (int e = t; e < EPER_; e += blockDim.x) {
    int s = esrc[e], d = edst[e], r = erel[e];
    float inv = 1.0f / fmaxf((float)cnt[d * R_ + r], 1.0f);
    uint2 p;
    p.x = (unsigned)s | ((unsigned)d << 8) | ((unsigned)r << 16);
    p.y = __float_as_uint(inv);
    ws_edges[e] = p;
  }
  for (int i = t; i < R_ * H_ * FEAT_; i += blockDim.x) {
    int r = i / (H_ * FEAT_);
    int rem = i % (H_ * FEAT_);
    int h = rem / FEAT_, k = rem % FEAT_;
    wrelT[i] = (_Float16)w_rel[(r * FEAT_ + k) * H_ + h];
  }
  for (int i = t; i < H_ * FEAT_; i += blockDim.x) {
    int h = i / FEAT_, k = i % FEAT_;
    wrootT[i] = (_Float16)w_root[k * H_ + h];
  }
}

// LDS layout offsets (bytes)
#define OFF_X     0                       // f32 x, double buffered: 2*8192
#define OFF_HREL  16384                   // 6 * 64*32 f32 = 49152
#define OFF_AGG   (16384 + 49152)         // 8192
#define OFF_EDGE  (16384 + 49152 + 8192)  // 12288
#define OFF_PART  (16384 + 49152 + 8192 + 12288)   // 1024
#define OFF_TAIL  (OFF_PART + 1024)                // cin(256)+mid(128)+allq(64)
#define SMEM_SZ   (OFF_TAIL + 448)

// ---------------------------------------------------------------------------
// Fused kernel: one block per graph, 8 waves, loops over 3 agents.
// TDM double-buffers the next agent's x tile (HBM->LDS DMA) while the current
// agent runs GEMM (v_wmma_f32_16x16x32_f16, one WMMA per 16x16 tile, K=32),
// ds_add_f32 edge aggregation, maxpool and the critic head.
// ---------------------------------------------------------------------------
__global__ void __launch_bounds__(256) critic_kernel(
    const float* __restrict__ unary, const float* __restrict__ actions,
    const float* __restrict__ b_gnn, const float* __restrict__ w1,
    const float* __restrict__ b1, const float* __restrict__ w2,
    const float* __restrict__ b2,
    const uint2* __restrict__ ws_edges,
    const _Float16* __restrict__ wrelT, const _Float16* __restrict__ wrootT,
    float* __restrict__ out) {
  extern __shared__ char smem[];
  float*   xf32  = (float*)(smem + OFF_X);
  float*   hrel  = (float*)(smem + OFF_HREL);
  float*   agg   = (float*)(smem + OFF_AGG);
  uint2*   eLds  = (uint2*)(smem + OFF_EDGE);
  float* partial = (float*)(smem + OFF_PART);
  float* cin     = (float*)(smem + OFF_TAIL);
  float* mid     = cin + 64;
  float* allq    = mid + 32;

  const int t    = threadIdx.x;
  const int lane = t & 31;
  const int w    = t >> 5;
  const int b    = blockIdx.x;

  const unsigned ldsBase = (unsigned)(uintptr_t)(void*)smem;

  // Kick off DMA: agent-0 x tile and the shared edge table (wave 0 only;
  // TENSORcnt is per-wave, completion published via __syncthreads).
  if (w == 0) {
    tdm_load_1d(ldsBase + OFF_X,
                unary + (size_t)0 * B_ * N_ * FEAT_ + (size_t)b * N_ * FEAT_,
                N_ * FEAT_, 2u /*4B*/);
    tdm_load_1d(ldsBase + OFF_EDGE, ws_edges, EPER_, 3u /*8B*/);
  }

  // Wave -> 16x16 tile of the 64x32 GEMM output (4 M-tiles x 2 N-tiles)
  const int m0   = (w >> 1) * 16;
  const int n0   = (w & 1) * 16;
  const int rowA = m0 + (lane & 15);
  const int klo  = (lane < 16) ? 0 : 8;   // ISA 16-bit fragment k-split
  const int colB = n0 + (lane & 15);

  for (int a = 0; a < A_; ++a) {
    if (w == 0) __builtin_amdgcn_s_wait_tensorcnt(0);  // x[a] (+edges) landed
    __syncthreads();                                   // publish; prior iter done

    // Prefetch next agent's x into the other buffer while we compute.
    if (w == 0 && a + 1 < A_) {
      tdm_load_1d(ldsBase + OFF_X + ((a + 1) & 1) * (N_ * FEAT_ * 4),
                  unary + (size_t)(a + 1) * B_ * N_ * FEAT_ + (size_t)b * N_ * FEAT_,
                  N_ * FEAT_, 2u);
    }

    // Build A-fragment from f32 LDS (convert in-register), reused for 7 GEMMs.
    const float* xb = xf32 + (a & 1) * (N_ * FEAT_);
    float tf[16];
    *(float4*)&tf[0]  = *(const float4*)&xb[rowA * FEAT_ + klo];
    *(float4*)&tf[4]  = *(const float4*)&xb[rowA * FEAT_ + klo + 4];
    *(float4*)&tf[8]  = *(const float4*)&xb[rowA * FEAT_ + klo + 16];
    *(float4*)&tf[12] = *(const float4*)&xb[rowA * FEAT_ + klo + 20];
    FragAB fa;
#pragma unroll
    for (int i = 0; i < 16; ++i) fa.v[i] = (_Float16)tf[i];

    // 6 relation GEMMs + root GEMM (root written straight into agg).
    for (int g = 0; g < R_ + 1; ++g) {
      const _Float16* wt = (g < R_) ? (wrelT + g * (H_ * FEAT_)) : wrootT;
      FragAB fb;
      fb.q[0] = *(const uint4*)&wt[colB * FEAT_ + klo];
      fb.q[1] = *(const uint4*)&wt[colB * FEAT_ + klo + 16];
      v8f c = {};
      c = __builtin_amdgcn_wmma_f32_16x16x32_f16(false, fa.v, false, fb.v,
                                                 (short)0, c, false, false);
      FragC fc; fc.v = c;
      float* dst = (g < R_) ? (hrel + g * (N_ * H_)) : agg;
      const int rb = m0 + ((lane < 16) ? 0 : 8);
#pragma unroll
      for (int i = 0; i < 8; ++i) dst[(rb + i) * H_ + colB] = fc.f[i];
    }
    __syncthreads();

    // Edge aggregation: wave per edge, lane per channel, ds_add_f32 atomics.
    for (int e = w; e < EPER_; e += 8) {
      uint2 ep = eLds[e];
      int s = ep.x & 255, d = (ep.x >> 8) & 255, r = (ep.x >> 16) & 255;
      float norm = __uint_as_float(ep.y);
      float val = hrel[r * (N_ * H_) + s * H_ + lane] * norm;
      atomicAdd(&agg[d * H_ + lane], val);
    }
    __syncthreads();

    // relu(agg + b_gnn) + partial max-pool over nodes (8 groups of 8 rows).
    {
      int c = t & 31, g = t >> 5;
      float bg = b_gnn[c];
      float m = 0.0f;  // relu outputs >= 0: 0 is a valid identity
      for (int r = g * 8; r < g * 8 + 8; ++r) {
        float v = fmaxf(agg[r * H_ + c] + bg, 0.0f);
        m = fmaxf(m, v);
      }
      partial[g * 32 + c] = m;
    }
    __syncthreads();

    // Final pool + critic input assembly.
    if (t < 32) {
      float m = partial[t];
      for (int g = 1; g < 8; ++g) m = fmaxf(m, partial[g * 32 + t]);
      cin[t] = m;
    } else if (t < 64) {
      int i2 = t - 32;
      int slot = i2 >> 4, k = i2 & 15;
      int ja;
      if (a == 0)      ja = (slot == 0) ? 1 : 2;
      else if (a == 1) ja = (slot == 0) ? 0 : 2;
      else             ja = (slot == 0) ? 0 : 1;
      cin[t] = actions[((size_t)ja * B_ + b) * NACT_ + k];
    }
    __syncthreads();

    // mid = leaky_relu(cin @ w1[a] + b1[a]) : 64x32 matvec, lane per channel.
    if (t < 32) {
      float s = b1[a * H_ + t];
      const float* w1a = w1 + (size_t)a * (H_ + NACT_ * (A_ - 1)) * H_;
      for (int i = 0; i < 64; ++i) s += cin[i] * w1a[i * H_ + t];
      mid[t] = (s > 0.0f) ? s : 0.01f * s;
    }
    __syncthreads();

    // all_q = mid @ w2[a] + b2[a]
    if (t < 16) {
      float s = b2[a * NACT_ + t];
      const float* w2a = w2 + (size_t)a * H_ * NACT_;
      for (int h = 0; h < H_; ++h) s += mid[h] * w2a[h * NACT_ + t];
      allq[t] = s;
    }
    __syncthreads();

    // Select q at argmax(actions[a]) (first-max semantics like jnp.argmax).
    if (t == 0) {
      const float* act = actions + ((size_t)a * B_ + b) * NACT_;
      int best = 0; float bv = act[0];
      for (int j = 1; j < NACT_; ++j)
        if (act[j] > bv) { bv = act[j]; best = j; }
      out[a * B_ + b] = allq[best];
    }
  }
}

// ---------------------------------------------------------------------------
extern "C" void kernel_launch(void* const* d_in, const int* in_sizes, int n_in,
                              void* d_out, int out_size, void* d_ws, size_t ws_size,
                              hipStream_t stream) {
  const float* unary   = (const float*)d_in[0];
  const float* actions = (const float*)d_in[1];
  const float* w_rel   = (const float*)d_in[2];
  const float* w_root  = (const float*)d_in[3];
  const float* b_gnn   = (const float*)d_in[4];
  const float* w1      = (const float*)d_in[5];
  const float* b1      = (const float*)d_in[6];
  const float* w2      = (const float*)d_in[7];
  const float* b2      = (const float*)d_in[8];
  const int*   esrc    = (const int*)d_in[9];
  const int*   edst    = (const int*)d_in[10];
  const int*   erel    = (const int*)d_in[11];
  float* out = (float*)d_out;

  char* ws = (char*)d_ws;
  uint2*    ws_edges = (uint2*)ws;                 // 1536 * 8   = 12288 B
  _Float16* wrelT    = (_Float16*)(ws + 12288);    // 6144 * 2   = 12288 B
  _Float16* wrootT   = (_Float16*)(ws + 24576);    // 1024 * 2   =  2048 B

  prep_kernel<<<1, 256, 0, stream>>>(esrc, edst, erel, w_rel, w_root,
                                     ws_edges, wrelT, wrootT);

  critic_kernel<<<B_, 256, SMEM_SZ, stream>>>(unary, actions, b_gnn, w1, b1,
                                              w2, b2, ws_edges, wrelT, wrootT,
                                              out);
}
